// MoLoRA10M_16690242912264
// MI455X (gfx1250) — matrix operand order repaired and checked by reading, not compiled, not run on hardware
//
#include <hip/hip_runtime.h>
#include <hip/hip_bf16.h>
#include <cstdint>

#define H      640
#define RH     256
#define NE     5
#define RANK   8
#define ER     40     // NE*RANK
#define ERP    48     // padded to 3 WMMA N-tiles
#define KPAD   64     // final GEMM K padded (40 -> 64, two k-steps of 32)
#define SCALING 2.0f  // 16.0 / RANK

typedef __attribute__((ext_vector_type(16))) __bf16          v16bf;
typedef __attribute__((ext_vector_type(8)))  float           v8f;
typedef __attribute__((ext_vector_type(4)))  int             v4i;

#define AS1 __attribute__((address_space(1)))
#define AS3 __attribute__((address_space(3)))

#if __has_builtin(__builtin_amdgcn_global_load_async_to_lds_b128) && \
    __has_builtin(__builtin_amdgcn_s_wait_asynccnt)
#define USE_ASYNC_LDS 1
#else
#define USE_ASYNC_LDS 0
#endif

__device__ __forceinline__ unsigned short f2bf(float f) {
  return __builtin_bit_cast(unsigned short, static_cast<__bf16>(f));
}
__device__ __forceinline__ float bf2f(unsigned short b) {
  return __builtin_bit_cast(float, (uint32_t)b << 16);
}

// ---------------------------------------------------------------------------
// Weight prep: W1 -> bf16; A[E,H,R] -> A_all[640x48] bf16 (cols 40..47 = 0);
// Bm[E,R,H] -> Bm_pad[64x640] bf16 (rows 40..63 = 0).
// ---------------------------------------------------------------------------
__global__ void molora_prep(const float* __restrict__ W1,
                            const float* __restrict__ A,
                            const float* __restrict__ Bm,
                            unsigned short* __restrict__ w1bf,
                            unsigned short* __restrict__ aall,
                            unsigned short* __restrict__ bmp) {
  const int i = blockIdx.x * 256 + threadIdx.x;
  if (i < H * RH) w1bf[i] = f2bf(W1[i]);
  if (i < H * ERP) {
    const int k = i / ERP, j = i % ERP;
    aall[i] = (j < ER) ? f2bf(A[((size_t)(j >> 3) * H + k) * RANK + (j & 7)])
                       : (unsigned short)0;
  }
  if (i < KPAD * H) {
    const int j = i / H, hh = i % H;
    bmp[i] = (j < ER) ? f2bf(Bm[((size_t)(j >> 3) * RANK + (j & 7)) * H + hh])
                      : (unsigned short)0;
  }
}

// ---------------------------------------------------------------------------
// Fused MoLoRA kernel: 128 tokens / block, 8 wave32 / block, 16 tokens / wave.
// ---------------------------------------------------------------------------
__global__ __launch_bounds__(256)
void molora_main(const float* __restrict__ x,
                 const float* __restrict__ base,
                 const float* __restrict__ b1,
                 const float* __restrict__ W2,
                 const float* __restrict__ b2,
                 const unsigned short* __restrict__ w1bf,
                 const unsigned short* __restrict__ aall,
                 const unsigned short* __restrict__ bmp,
                 float* __restrict__ out) {
  // LDS carve-out (73 KB):
  //  [0, 65536):  phase A: W1/A chunk stage (32 x 304 bf16 = 19456 B)
  //               phase B: h_lds 128 x 256 bf16 (65536 B)
  //               phase C: v_lds 128 x 64 bf16  (16384 B)
  //  [65536, 68096): w_lds  128 x 5 f32 (gate weights, SCALING folded)
  //  [68096, 73216): w2_lds 256 x 5 f32
  //  [73216, 74240): b1_lds 256 f32
  //  [74240, 74272): b2_lds 5 f32
  __shared__ __attribute__((aligned(32))) unsigned char smem[74272];
  unsigned short* lds_bf = (unsigned short*)smem;
  float* w_lds  = (float*)(smem + 65536);
  float* w2_lds = (float*)(smem + 68096);
  float* b1_lds = (float*)(smem + 73216);
  float* b2_lds = (float*)(smem + 74240);

  const int tid   = threadIdx.x;
  const int lane  = tid & 31;
  const int wave  = tid >> 5;
  const int rowg0 = blockIdx.x * 128;   // first global token of this block
  const int rowl0 = wave * 16;          // wave's first local row

  // Stage router small params (covered by first barrier in the k-loop).
  for (int i = tid; i < RH * NE; i += 256) w2_lds[i] = W2[i];
  if (tid < RH) b1_lds[tid] = b1[tid];
  if (tid < NE) b2_lds[tid] = b2[tid];

  const v8f vzero = {0.f, 0.f, 0.f, 0.f, 0.f, 0.f, 0.f, 0.f};
  v8f hacc[16];   // router hidden: 16 rows x 256 cols (16 N-tiles)
  v8f lacc[3];    // LoRA low:      16 rows x 48  cols (3  N-tiles)
#pragma unroll
  for (int i = 0; i < 16; ++i) hacc[i] = vzero;
#pragma unroll
  for (int i = 0; i < 3; ++i) lacc[i] = vzero;

  // A-fragment addressing (16x32 bf16 layout, CDNA5 ISA 7.12.2):
  // lane<16: rows 0..15, k in [0,8)+[16,24); lane>=16: same rows, k +8.
  const int arow = rowg0 + rowl0 + (lane & 15);
  const int alok = (lane >> 4) << 3;

  // ---------------- Phase A: h = x@W1, low = x@A_all (x read once) ---------
  for (int kc = 0; kc < H / 32; ++kc) {
    const int k0 = kc * 32;
    __syncthreads();
#if USE_ASYNC_LDS
    {  // async DMA stage: W1 chunk [32x256] + A_all chunk [32x48], stride 304
      for (int u = tid; u < 32 * 32; u += 256) {          // 16B units, W1 part
        const int k = u >> 5, o = u & 31;
        const unsigned short* g = w1bf + (size_t)(k0 + k) * RH + o * 8;
        unsigned short* l = lds_bf + k * 304 + o * 8;
        __builtin_amdgcn_global_load_async_to_lds_b128(
            (AS1 v4i*)(uintptr_t)g,
            (AS3 v4i*)(uint32_t)(uintptr_t)l, 0, 0);
      }
      if (tid < 32 * 6) {                                 // 16B units, A part
        const int k = tid / 6, o = tid % 6;
        const unsigned short* g = aall + (size_t)(k0 + k) * ERP + o * 8;
        unsigned short* l = lds_bf + k * 304 + RH + o * 8;
        __builtin_amdgcn_global_load_async_to_lds_b128(
            (AS1 v4i*)(uintptr_t)g,
            (AS3 v4i*)(uint32_t)(uintptr_t)l, 0, 0);
      }
      __builtin_amdgcn_s_wait_asynccnt(0);
    }
#else
    {  // fallback: VGPR-copy stage
      const uint32_t* s32 = (const uint32_t*)(w1bf + (size_t)k0 * RH);
      uint32_t* d32 = (uint32_t*)lds_bf;
      for (int i = tid; i < 32 * 128; i += 256) {
        const int k = i >> 7, n = i & 127;
        d32[k * 152 + n] = s32[i];
      }
      const uint32_t* a32 = (const uint32_t*)(aall + (size_t)k0 * ERP);
      for (int i = tid; i < 32 * 24; i += 256) {
        const int k = i / 24, n = i % 24;
        d32[k * 152 + 128 + n] = a32[i];
      }
    }
#endif
    __syncthreads();

    // Per-lane x fragment: two contiguous 8-float runs, fp32 -> bf16 (native cvt).
    const float4* xr = (const float4*)(x + (size_t)arow * H + k0 + alok);
    const float4 r0 = xr[0], r1 = xr[1];
    const float4 r2 = xr[4], r3 = xr[5];  // +16 floats
    if (kc + 1 < H / 32)
      __builtin_prefetch(x + (size_t)arow * H + k0 + 32 + alok, 0, 0);
    v16bf afrag;
    afrag[0]  = (__bf16)r0.x; afrag[1]  = (__bf16)r0.y;
    afrag[2]  = (__bf16)r0.z; afrag[3]  = (__bf16)r0.w;
    afrag[4]  = (__bf16)r1.x; afrag[5]  = (__bf16)r1.y;
    afrag[6]  = (__bf16)r1.z; afrag[7]  = (__bf16)r1.w;
    afrag[8]  = (__bf16)r2.x; afrag[9]  = (__bf16)r2.y;
    afrag[10] = (__bf16)r2.z; afrag[11] = (__bf16)r2.w;
    afrag[12] = (__bf16)r3.x; afrag[13] = (__bf16)r3.y;
    afrag[14] = (__bf16)r3.z; afrag[15] = (__bf16)r3.w;

    // B fragment: lane%16 = N-col, k = 8h+s / 16+8h+(s-8) over elements.
    const unsigned short* brow = lds_bf + (lane & 15);
    const int bk = (lane >> 4) << 3;
#pragma unroll
    for (int nt = 0; nt < 16; ++nt) {
      v16bf bfrag;
#pragma unroll
      for (int i = 0; i < 8; ++i) {
        bfrag[i]     = __builtin_bit_cast(__bf16, brow[(bk + i) * 304 + nt * 16]);
        bfrag[8 + i] = __builtin_bit_cast(__bf16, brow[(bk + 16 + i) * 304 + nt * 16]);
      }
      hacc[nt] = __builtin_amdgcn_wmma_f32_16x16x32_bf16(
          false, afrag, false, bfrag, (short)0, hacc[nt], false, false);
    }
#pragma unroll
    for (int j = 0; j < 3; ++j) {
      v16bf bfrag;
#pragma unroll
      for (int i = 0; i < 8; ++i) {
        bfrag[i]     = __builtin_bit_cast(__bf16, brow[(bk + i) * 304 + RH + j * 16]);
        bfrag[8 + i] = __builtin_bit_cast(__bf16, brow[(bk + 16 + i) * 304 + RH + j * 16]);
      }
      lacc[j] = __builtin_amdgcn_wmma_f32_16x16x32_bf16(
          false, afrag, false, bfrag, (short)0, lacc[j], false, false);
    }
  }

  // ---------------- Phase B: SiLU -> h_lds; router logits/top-2 ------------
  __syncthreads();  // staging dead; region becomes h_lds
  {
    const int rbase = rowl0 + ((lane >> 4) << 3);  // C layout: row = j + 8*(l>>4)
    const int cl = lane & 15;
#pragma unroll
    for (int nt = 0; nt < 16; ++nt) {
      const int col = nt * 16 + cl;
      const float bb = b1_lds[col];
#pragma unroll
      for (int j = 0; j < 8; ++j) {
        float v = hacc[nt][j] + bb;
        v = v / (1.0f + __expf(-v));  // SiLU
        lds_bf[(rbase + j) * RH + col] = f2bf(v);
      }
    }
  }
  __syncthreads();

  if (tid < 128) {  // one thread per token: 256x5 logits, softmax, top-2
    float l0 = b2_lds[0], l1 = b2_lds[1], l2 = b2_lds[2], l3 = b2_lds[3], l4 = b2_lds[4];
    const uint4* h128 = (const uint4*)(lds_bf + tid * RH);
    for (int k8 = 0; k8 < RH / 8; ++k8) {
      const uint4 q = h128[k8];               // 8 bf16
      float hv[8];
      hv[0] = bf2f((unsigned short)(q.x & 0xFFFF));
      hv[1] = bf2f((unsigned short)(q.x >> 16));
      hv[2] = bf2f((unsigned short)(q.y & 0xFFFF));
      hv[3] = bf2f((unsigned short)(q.y >> 16));
      hv[4] = bf2f((unsigned short)(q.z & 0xFFFF));
      hv[5] = bf2f((unsigned short)(q.z >> 16));
      hv[6] = bf2f((unsigned short)(q.w & 0xFFFF));
      hv[7] = bf2f((unsigned short)(q.w >> 16));
#pragma unroll
      for (int j = 0; j < 8; ++j) {
        const float* wr = w2_lds + (k8 * 8 + j) * NE;
        l0 += hv[j] * wr[0]; l1 += hv[j] * wr[1]; l2 += hv[j] * wr[2];
        l3 += hv[j] * wr[3]; l4 += hv[j] * wr[4];
      }
    }
    float lg[NE] = {l0, l1, l2, l3, l4};
    float m = lg[0];
#pragma unroll
    for (int e = 1; e < NE; ++e) m = fmaxf(m, lg[e]);
    float p[NE];
#pragma unroll
    for (int e = 0; e < NE; ++e) p[e] = __expf(lg[e] - m);
    int i0 = 0;
#pragma unroll
    for (int e = 1; e < NE; ++e) if (p[e] > p[i0]) i0 = e;
    int i1 = (i0 == 0) ? 1 : 0;
#pragma unroll
    for (int e = 0; e < NE; ++e) if (e != i1 && e != i0 && p[e] > p[i1]) i1 = e;
    const float inv = SCALING / (p[i0] + p[i1]);  // softmax denom cancels
    float* wr = w_lds + tid * NE;
#pragma unroll
    for (int e = 0; e < NE; ++e)
      wr[e] = (e == i0) ? p[i0] * inv : (e == i1) ? p[i1] * inv : 0.f;
  }
  __syncthreads();

  // ---------------- Phase C: gate*low -> v_lds; out = base + v @ Bm_pad ----
  unsigned short* v_lds = lds_bf;  // 128 x 64 bf16 (h_lds dead)
  {
    const int rb = rowl0 + ((lane >> 4) << 3);
    const int cl = lane & 15;
#pragma unroll
    for (int j3 = 0; j3 < 3; ++j3) {
      const int c = j3 * 16 + cl;
#pragma unroll
      for (int j = 0; j < 8; ++j) {
        const int r = rb + j;
        const float w = (c < ER) ? w_lds[r * NE + (c >> 3)] : 0.f;
        v_lds[r * KPAD + c] = f2bf(lacc[j3][j] * w);
      }
    }
#pragma unroll
    for (int j = 0; j < 8; ++j) v_lds[(rb + j) * KPAD + 48 + cl] = 0;
  }
  __syncthreads();

  v16bf afin[2];  // two k-steps of the 16x64 A matrix
  {
    const unsigned short* vr = v_lds + (rowl0 + (lane & 15)) * KPAD;
    const int kb = (lane >> 4) << 3;
#pragma unroll
    for (int kt = 0; kt < 2; ++kt) {
      v16bf tt;
#pragma unroll
      for (int i = 0; i < 8; ++i) {
        tt[i]     = __builtin_bit_cast(__bf16, vr[kt * 32 + kb + i]);
        tt[8 + i] = __builtin_bit_cast(__bf16, vr[kt * 32 + kb + 16 + i]);
      }
      afin[kt] = tt;
    }
  }
  const int cl = lane & 15;
  const int rg = rowg0 + rowl0 + ((lane >> 4) << 3);
  for (int nt = 0; nt < H / 16; ++nt) {  // 40 output N-tiles
    const int n0 = nt * 16;
    v16bf bf0, bf1;
#pragma unroll
    for (int i = 0; i < 8; ++i) {
      const int kk = ((lane >> 4) << 3) + i;
      bf0[i]     = __builtin_bit_cast(__bf16, bmp[(size_t)kk * H + n0 + cl]);
      bf0[8 + i] = __builtin_bit_cast(__bf16, bmp[(size_t)(kk + 16) * H + n0 + cl]);
      bf1[i]     = __builtin_bit_cast(__bf16, bmp[(size_t)(32 + kk) * H + n0 + cl]);
      bf1[8 + i] = __builtin_bit_cast(__bf16, bmp[(size_t)(48 + kk) * H + n0 + cl]);
    }
    v8f acc = vzero;
    acc = __builtin_amdgcn_wmma_f32_16x16x32_bf16(false, afin[0], false, bf0,
                                                  (short)0, acc, false, false);
    acc = __builtin_amdgcn_wmma_f32_16x16x32_bf16(false, afin[1], false, bf1,
                                                  (short)0, acc, false, false);
    const int colg = n0 + cl;
#pragma unroll
    for (int j = 0; j < 8; ++j) {
      const size_t off = (size_t)(rg + j) * H + colg;
      out[off] = base[off] + acc[j];
    }
  }
}

// ---------------------------------------------------------------------------
extern "C" void kernel_launch(void* const* d_in, const int* in_sizes, int n_in,
                              void* d_out, int out_size, void* d_ws, size_t ws_size,
                              hipStream_t stream) {
  const float* x  = (const float*)d_in[0];
  const float* bo = (const float*)d_in[1];
  const float* W1 = (const float*)d_in[2];
  const float* b1 = (const float*)d_in[3];
  const float* W2 = (const float*)d_in[4];
  const float* b2 = (const float*)d_in[5];
  const float* A  = (const float*)d_in[6];
  const float* Bm = (const float*)d_in[7];
  float* out = (float*)d_out;
  const int NT = in_sizes[0] / H;  // 32768 tokens

  unsigned short* w1bf = (unsigned short*)d_ws;                      // 640*256*2 B
  unsigned short* aall = (unsigned short*)((char*)d_ws + 327680);    // 640*48*2 B
  unsigned short* bmp  = (unsigned short*)((char*)d_ws + 389120);    // 64*640*2 B

  molora_prep<<<640, 256, 0, stream>>>(W1, A, Bm, w1bf, aall, bmp);
  molora_main<<<NT / 128, 256, 0, stream>>>(x, bo, b1, W2, b2, w1bf, aall, bmp, out);
}
